// VectorQuantizer_76897094468398
// MI455X (gfx1250) — matrix operand order repaired and checked by reading, not compiled, not run on hardware
//
#include <hip/hip_runtime.h>
#include <stdint.h>

// ---------------- problem constants ----------------
#define N_TOKENS   131072
#define N_EMBED    2048
#define EMBED_DIM  256
#define CHUNK      64                      // codes staged in LDS per step
#define NCHUNK     (N_EMBED / CHUNK)       // 32
#define ROWBYTES   (EMBED_DIM * 2)         // 512 B per f16 codebook row

typedef __attribute__((ext_vector_type(16))) _Float16 v16h;
typedef __attribute__((ext_vector_type(8)))  _Float16 v8h;
typedef __attribute__((ext_vector_type(8)))  float    v8f;

// =====================================================================
// Kernel 1: convert codebook to f16 + compute ||w_k||^2 (fp32 exact).
// =====================================================================
__global__ __launch_bounds__(256)
void vq_prep(const float* __restrict__ w, _Float16* __restrict__ wf16,
             float* __restrict__ wnorm) {
    const int c = blockIdx.x;
    const int t = threadIdx.x;
    const float v = w[(size_t)c * EMBED_DIM + t];
    wf16[(size_t)c * EMBED_DIM + t] = (_Float16)v;
    float s = v * v;
    #pragma unroll
    for (int m = 16; m >= 1; m >>= 1) s += __shfl_xor(s, m, 32);
    __shared__ float ps[8];
    if ((t & 31) == 0) ps[t >> 5] = s;
    __syncthreads();
    if (t == 0) {
        float tot = 0.f;
        #pragma unroll
        for (int i = 0; i < 8; ++i) tot += ps[i];
        wnorm[c] = tot;
    }
}

// =====================================================================
// Kernel 2: fused WMMA score GEMM + argmin.
// Block = 128 threads (4 waves), 64 tokens/block (16 per wave).
// A (z, f16 hi+lo) in registers; B (codebook f16) double-buffered
// through LDS with async-to-LDS copies, XOR-swizzled against bank
// conflicts. One B copy per block instead of one per wave.
// =====================================================================
__global__ __launch_bounds__(128)
void vq_gemm_argmin(const float* __restrict__ z,
                    const _Float16* __restrict__ wf16,
                    const float* __restrict__ wnorm,
                    int* __restrict__ code_out) {
    __shared__ __align__(16) char swb[2][CHUNK * ROWBYTES];   // 2 x 32 KB

    const int tid  = threadIdx.x;
    const int wave = tid >> 5;
    const int lane = tid & 31;
    const size_t tok0 = (size_t)blockIdx.x * 64;

    // ---- build A fragments straight from global (one-time per block) ----
    // ISA 16-bit A 16x32 layout: lanes 0-15 (token=lane): K {0-7, 16-23};
    // lanes 16-31 (token=lane-16): K {8-15, 24-31}.
    const int mrow  = lane & 15;
    const int khalf = (lane >> 4) ? 8 : 0;
    const float* myrow = z + (tok0 + (size_t)(wave * 16 + mrow)) * EMBED_DIM;

    v16h a_hi[8], a_lo[8];
    #pragma unroll
    for (int kc = 0; kc < 8; ++kc) {
        const float* p0 = myrow + kc * 32 + khalf;
        const float* p1 = myrow + kc * 32 + 16 + khalf;
        #pragma unroll
        for (int j = 0; j < 8; ++j) {
            const float f0 = p0[j];
            const _Float16 h0 = (_Float16)f0;
            a_hi[kc][j]     = h0;
            a_lo[kc][j]     = (_Float16)(f0 - (float)h0);
            const float f1 = p1[j];
            const _Float16 h1 = (_Float16)f1;
            a_hi[kc][8 + j] = h1;
            a_lo[kc][8 + j] = (_Float16)(f1 - (float)h1);
        }
    }

    // ---- async W-chunk copier: 64 rows x 512 B, XOR-swizzled 16B slots ----
    const char* wbytes = (const char*)wf16;
    const unsigned lb[2] = { (unsigned)(uintptr_t)(&swb[0][0]),
                             (unsigned)(uintptr_t)(&swb[1][0]) };

    auto issue_chunk = [&](int cb, unsigned lbase) {
        const size_t gbase = (size_t)cb * CHUNK * ROWBYTES;
        #pragma unroll
        for (int i = 0; i < 16; ++i) {                 // 16 async instr / wave
            const int id    = i * 128 + tid;           // [0, 2048)
            const int row   = id >> 5;                 // 32 x 16B per row
            const int col16 = id & 31;
            const unsigned loff = lbase +
                (unsigned)(row * ROWBYTES + ((col16 ^ (row & 31)) << 4));
            const unsigned long long ga = (unsigned long long)(uintptr_t)
                (wbytes + gbase + (size_t)row * ROWBYTES + (size_t)col16 * 16);
            asm volatile("global_load_async_to_lds_b128 %0, %1, off"
                         :: "v"(loff), "v"(ga) : "memory");
        }
    };

    // ---- sweep all codes: double-buffered LDS staging ----
    // B 32x16 layout: lane n (0-15) = col n, K {0-7, 16-23}; lane 16+n = K {8-15, 24-31}
    // => 2 x 16B swizzled LDS slots per lane per k-step.
    const int ncol = lane & 15;
    const int c0   = (lane >> 4) ? 1 : 0;

    float minv[8];
    int   mini[8];
    #pragma unroll
    for (int r = 0; r < 8; ++r) { minv[r] = 3.4e38f; mini[r] = 0; }

    issue_chunk(0, lb[0]);
    for (int cb = 0; cb < NCHUNK; ++cb) {
        const int b = cb & 1;
        if (cb + 1 < NCHUNK) {
            issue_chunk(cb + 1, lb[b ^ 1]);
            asm volatile("s_wait_asynccnt 0x10" ::: "memory"); // chunk cb done
        } else {
            asm volatile("s_wait_asynccnt 0x0" ::: "memory");
        }
        __syncthreads();                       // everyone's copies visible

        #pragma unroll
        for (int ncl = 0; ncl < 4; ++ncl) {
            const int row = ncl * 16 + ncol;   // code row inside chunk
            const int xr  = row & 31;
            const int codecol = cb * CHUNK + row;

            v8f acc = {};
            #pragma unroll
            for (int kc = 0; kc < 8; ++kc) {
                const int c = kc * 4 + c0;     // 16B slot index
                const v8h b0 = *(const v8h*)
                    &swb[b][row * ROWBYTES + ((c ^ xr) << 4)];
                const v8h b1 = *(const v8h*)
                    &swb[b][row * ROWBYTES + (((c + 2) ^ xr) << 4)];
                const v16h bb = __builtin_shufflevector(
                    b0, b1, 0, 1, 2, 3, 4, 5, 6, 7,
                    8, 9, 10, 11, 12, 13, 14, 15);
                acc = __builtin_amdgcn_wmma_f32_16x16x32_f16(
                    false, a_hi[kc], false, bb, (short)0, acc, false, false);
                acc = __builtin_amdgcn_wmma_f32_16x16x32_f16(
                    false, a_lo[kc], false, bb, (short)0, acc, false, false);
            }

            const float wn = wnorm[codecol];
            #pragma unroll
            for (int r = 0; r < 8; ++r) {
                const float dist = wn - 2.0f * acc[r];  // ||z||^2 row-const: dropped
                if (dist < minv[r]) { minv[r] = dist; mini[r] = codecol; }
            }
        }
        __syncthreads();   // all waves done reading buf b before it refills
    }

    // ---- reduce argmin across the 16 lanes holding the same token row ----
    // C layout: VGPR r, lanes 0-15 -> M=r ; lanes 16-31 -> M=8+r.
    #pragma unroll
    for (int r = 0; r < 8; ++r) {
        float v = minv[r];
        int   i = mini[r];
        #pragma unroll
        for (int m = 8; m >= 1; m >>= 1) {  // xor masks stay in 16-lane halves
            const float ov = __shfl_xor(v, m, 32);
            const int   oi = __shfl_xor(i, m, 32);
            if (ov < v || (ov == v && oi < i)) { v = ov; i = oi; }
        }
        if ((lane & 15) == 0) {
            const int mtok = r + ((lane >> 4) ? 8 : 0);
            code_out[tok0 + wave * 16 + mtok] = i;
        }
    }
}

// =====================================================================
// Kernel 3: gather z_q (exact fp32) + per-block partial loss sums.
// One wave per token; block = 256 threads -> 8 tokens/block.
// =====================================================================
__global__ __launch_bounds__(256)
void vq_gather(const float* __restrict__ z, const float* __restrict__ w,
               const int* __restrict__ code, float* __restrict__ zq,
               float* __restrict__ partial) {
    const int wave = threadIdx.x >> 5;
    const int lane = threadIdx.x & 31;
    const size_t tok = (size_t)blockIdx.x * 8 + wave;

    const int c = code[tok];
    const float4* wr = (const float4*)(w + (size_t)c * EMBED_DIM);
    const float4* zr = (const float4*)(z + tok * EMBED_DIM);
    float4* qo = (float4*)(zq + tok * EMBED_DIM);

    float s = 0.f;
    #pragma unroll
    for (int i = 0; i < 2; ++i) {
        const int idx = lane + i * 32;
        const float4 wv = wr[idx];
        const float4 zv = zr[idx];
        qo[idx] = wv;
        const float dx = zv.x - wv.x, dy = zv.y - wv.y;
        const float dz = zv.z - wv.z, dw = zv.w - wv.w;
        s += dx * dx + dy * dy + dz * dz + dw * dw;
    }
    #pragma unroll
    for (int m = 16; m >= 1; m >>= 1) s += __shfl_xor(s, m, 32);
    __shared__ float ps[8];
    if (lane == 0) ps[wave] = s;
    __syncthreads();
    if (threadIdx.x == 0) {
        float tot = 0.f;
        #pragma unroll
        for (int i = 0; i < 8; ++i) tot += ps[i];
        partial[blockIdx.x] = tot;
    }
}

// =====================================================================
// Kernel 4: deterministic final loss reduction.
// =====================================================================
__global__ __launch_bounds__(256)
void vq_loss(const float* __restrict__ partial, float* __restrict__ loss) {
    __shared__ float ps[256];
    float s = 0.f;
    for (int i = threadIdx.x; i < N_TOKENS / 8; i += 256) s += partial[i];
    ps[threadIdx.x] = s;
    __syncthreads();
    for (int off = 128; off >= 1; off >>= 1) {
        if (threadIdx.x < (unsigned)off) ps[threadIdx.x] += ps[threadIdx.x + off];
        __syncthreads();
    }
    if (threadIdx.x == 0)
        *loss = 1.25f * ps[0] / (float)((double)N_TOKENS * (double)EMBED_DIM);
}

// =====================================================================
extern "C" void kernel_launch(void* const* d_in, const int* in_sizes, int n_in,
                              void* d_out, int out_size, void* d_ws, size_t ws_size,
                              hipStream_t stream) {
    const float* z = (const float*)d_in[0];          // [131072, 256] f32
    const float* w = (const float*)d_in[1];          // [2048, 256]   f32

    // output layout: [N*D f32 z_q][1 f32 loss][N int32 code]
    float* zq   = (float*)d_out;
    float* loss = zq + (size_t)N_TOKENS * EMBED_DIM;
    int*   code = (int*)(loss + 1);

    // workspace layout
    _Float16* wf16  = (_Float16*)d_ws;                                 // 1 MB
    float* wnorm    = (float*)((char*)d_ws + (size_t)N_EMBED * EMBED_DIM * 2);
    float* partial  = wnorm + N_EMBED;                                 // 64 KB

    vq_prep<<<N_EMBED, 256, 0, stream>>>(w, wf16, wnorm);
    vq_gemm_argmin<<<N_TOKENS / 64, 128, 0, stream>>>(z, wf16, wnorm, code);
    vq_gather<<<N_TOKENS / 8, 256, 0, stream>>>(z, w, code, zq, partial);
    vq_loss<<<1, 256, 0, stream>>>(partial, loss);
}
